// LSTMBlock_77893526880394
// MI455X (gfx1250) — matrix operand order repaired and checked by reading, not compiled, not run on hardware
//
#include <hip/hip_runtime.h>

// ---------------------------------------------------------------------------
// Fused LSTM block for MI455X (gfx1250, wave32, WMMA + TDM double buffering).
//
//   g_t = [x_t | h] @ [W_ih*W_pre | W_hh]^T + (W_ih*b_pre + b_ih + b_hh)
//
// One v_wmma_f32_16x16x32_f16 per gate per timestep per 16-batch tile.
// A (16x32 f16):  K 0..11 = x channels, 12..15 = pad, 16..27 = h, 28..31 = pad
// B (32x16 f16):  packed per gate by prep kernel; pad rows are zero so any
//                 garbage in the A-pads contributes nothing (this also lets h
//                 stores run unpredicated: cols 12..15 only ever multiply 0).
// C (16x16 f32):  per-column bias splat.
// x staging: TENSOR_LOAD_TO_LDS, 3D tile (t=11, c=12, m=16), double buffered
//            behind compute with s_wait_tensorcnt.
// ---------------------------------------------------------------------------

typedef __attribute__((ext_vector_type(16))) _Float16 v16h;
typedef __attribute__((ext_vector_type(8)))  float    v8f;
typedef __attribute__((ext_vector_type(4)))  unsigned int v4u;
typedef __attribute__((ext_vector_type(8)))  int      v8i;
typedef __attribute__((ext_vector_type(4)))  int      v4i;

#define B_TOT    32768
#define C_CH     12
#define T_SEQ    99
#define HID      12
#define T_CHUNK  11
#define N_CHUNK  9                        // 9 * 11 = 99
#define WAVES    2
#define BLOCK    (WAVES * 32)
#define ROWS_W   16                       // batch rows per wave
#define XS_ROW   (C_CH * T_CHUNK)         // 132 floats per batch row [c][t]
#define XS_BUF   (ROWS_W * XS_ROW + 64)   // +pad: harmless c=12..15 overreads
#define HS_ROW   20                       // padded: conflict-free b128 reads
#define HS_WAVE  (ROWS_W * HS_ROW)

#if __has_builtin(__builtin_amdgcn_tensor_load_to_lds)
#define HAVE_TDM 1
#endif
#if __has_builtin(__builtin_amdgcn_tanhf)
#define HAVE_TANH 1
#endif

__device__ __forceinline__ float fexp2(float x) {
#if __has_builtin(__builtin_amdgcn_exp2f)
  return __builtin_amdgcn_exp2f(x);
#else
  return exp2f(x);
#endif
}
__device__ __forceinline__ float frcp(float x) {
#if __has_builtin(__builtin_amdgcn_rcpf)
  return __builtin_amdgcn_rcpf(x);
#else
  return 1.0f / x;
#endif
}
__device__ __forceinline__ float sigm_exp(float x) {
  return frcp(1.0f + fexp2(-1.4426950408889634f * x));
}
__device__ __forceinline__ float tanhx(float x) {
#ifdef HAVE_TANH
  return __builtin_amdgcn_tanhf(x);
#else
  return 2.0f * sigm_exp(2.0f * x) - 1.0f;
#endif
}
__device__ __forceinline__ float sigm(float x) {
#ifdef HAVE_TANH
  return 0.5f + 0.5f * __builtin_amdgcn_tanhf(0.5f * x);  // 1 TRANS + 2 VALU
#else
  return sigm_exp(x);
#endif
}

#ifdef HAVE_TDM
// Issue one TDM load: 3D tile  dim0=t (11 of 99), dim1=c (12), dim2=m (16),
// global strides 99 / 1188 elements, data_size=4B, into LDS at lds_off.
__device__ __forceinline__ void tdm_issue(unsigned lds_off, unsigned long long ga) {
  v4u g0;
  g0[0] = 1u;                                                   // count=1
  g0[1] = (unsigned)__builtin_amdgcn_readfirstlane((int)lds_off);
  g0[2] = (unsigned)__builtin_amdgcn_readfirstlane((int)(unsigned)(ga & 0xffffffffull));
  g0[3] = (unsigned)__builtin_amdgcn_readfirstlane(
              (int)((unsigned)((ga >> 32) & 0x01ffffffull) | 0x80000000u)); // type=2
  v8i g1;
  g1[0] = 0x00020000;                       // data_size=2 -> 4 bytes
  g1[1] = (int)((T_SEQ & 0xffff) << 16);    // tensor_dim0 lo16  (=99)
  g1[2] = (int)((C_CH & 0xffff) << 16);     // td0 hi16=0 | tensor_dim1 lo16 (=12)
  g1[3] = (int)(T_CHUNK << 16);             // td1 hi16=0 | tile_dim0 (=11)
  g1[4] = (int)(C_CH | (ROWS_W << 16));     // tile_dim1 (=12) | tile_dim2 (=16)
  g1[5] = T_SEQ;                            // tensor_dim0_stride lo32 (=99)
  g1[6] = (int)((1188u & 0xffffu) << 16);   // s0 hi16=0 | tensor_dim1_stride lo16
  g1[7] = 0;                                // tensor_dim1_stride hi32
  v4i g2;
  g2[0] = ROWS_W;                           // tensor_dim2 (=16)
  g2[1] = 0; g2[2] = 0; g2[3] = 0;          // tile_dim3=0 (unused)
  v4i g3;
  g3[0] = 0; g3[1] = 0; g3[2] = 0; g3[3] = 0;
#if defined(__clang_major__) && (__clang_major__ >= 23)
  v8i g4;
  g4[0]=0; g4[1]=0; g4[2]=0; g4[3]=0; g4[4]=0; g4[5]=0; g4[6]=0; g4[7]=0;
  __builtin_amdgcn_tensor_load_to_lds(g0, g1, g2, g3, g4, 0);
#else
  __builtin_amdgcn_tensor_load_to_lds(g0, g1, g2, g3, 0);
#endif
}
__device__ __forceinline__ void tdm_wait1() {
#if __has_builtin(__builtin_amdgcn_s_wait_tensorcnt)
  __builtin_amdgcn_s_wait_tensorcnt((unsigned short)1);
#else
  asm volatile("s_wait_tensorcnt 0x1" ::: "memory");
#endif
  asm volatile("" ::: "memory");
}
__device__ __forceinline__ void tdm_wait0() {
#if __has_builtin(__builtin_amdgcn_s_wait_tensorcnt)
  __builtin_amdgcn_s_wait_tensorcnt((unsigned short)0);
#else
  asm volatile("s_wait_tensorcnt 0x0" ::: "memory");
#endif
  asm volatile("" ::: "memory");
}
#endif  // HAVE_TDM

// ---------------------------------------------------------------------------
// Prep: fold W_pre into W_ih, pack B operands (i,f,g,o,out) + bias vectors.
//   B layout: lane l holds column n=l%16; half hh holds K=(hh<8?0:16)+(l>=16?8:0)+hh%8.
// ws: [5][32] v16h (5120 B) then [5][16] float biases (320 B)
// ---------------------------------------------------------------------------
__global__ void lstm_prep(const float* __restrict__ W_pre, const float* __restrict__ b_pre,
                          const float* __restrict__ W_ih,  const float* __restrict__ W_hh,
                          const float* __restrict__ b_ih,  const float* __restrict__ b_hh,
                          const float* __restrict__ W_out, const float* __restrict__ b_out,
                          void* __restrict__ ws) {
  int tid = threadIdx.x;
  _Float16* Bp  = (_Float16*)ws;
  float*    bia = (float*)((char*)ws + 5 * 32 * 16 * sizeof(_Float16));
  if (tid >= 5 * 32) return;
  int mat = tid / 32, lane = tid % 32;
  int n = lane & 15;
  int hiK = (lane >= 16) ? 16 : 0;
  for (int hh = 0; hh < 16; ++hh) {
    int K = (hh < 8) ? (hiK / 2 + hh) : (16 + hiK / 2 + (hh - 8));
    float v = 0.0f;
    if (n < HID) {
      if (mat < 4) {                       // gates i,f,g,o
        int row = mat * HID + n;
        if (K < C_CH)                      // x part: (W_ih @ W_pre)
          v = W_ih[row * 2 + 0] * W_pre[K] + W_ih[row * 2 + 1] * W_pre[C_CH + K];
        else if (K >= 16 && K < 16 + HID)  // recurrent part
          v = W_hh[row * HID + (K - 16)];
      } else {                             // output projection
        if (K < HID) v = W_out[n * HID + K];
      }
    }
    Bp[(mat * 32 + lane) * 16 + hh] = (_Float16)v;
  }
  if (hiK == 0) {
    float bv = 0.0f;
    if (n < HID) {
      if (mat < 4) {
        int row = mat * HID + n;
        bv = W_ih[row * 2 + 0] * b_pre[0] + W_ih[row * 2 + 1] * b_pre[1]
           + b_ih[row] + b_hh[row];
      } else {
        bv = b_out[n];
      }
    }
    bia[mat * 16 + n] = bv;
  }
}

// ---------------------------------------------------------------------------
// Main kernel: each wave owns 16 batch rows; no cross-wave communication,
// no barriers. x staged by TDM (double buffered) into [m][c][t] LDS tiles.
// ---------------------------------------------------------------------------
__global__ void __launch_bounds__(BLOCK)
lstm_main(const float* __restrict__ x, const void* __restrict__ ws,
          float* __restrict__ out) {
  __shared__ float xs[WAVES][2][XS_BUF];
  __shared__ float hs[WAVES][HS_WAVE];

  const int lane = threadIdx.x & 31;
  const int w    = threadIdx.x >> 5;
  const int b0   = blockIdx.x * (WAVES * ROWS_W) + w * ROWS_W;
  const int m    = lane & 15;          // batch row (A) / output column (D)
  const int hi   = lane >> 4;

  // Packed weights + biases
  const v16h* bp = (const v16h*)ws;
  const v16h Bi = bp[0 * 32 + lane];
  const v16h Bf = bp[1 * 32 + lane];
  const v16h Bg = bp[2 * 32 + lane];
  const v16h Bo = bp[3 * 32 + lane];
  const v16h Bz = bp[4 * 32 + lane];
  const float* bia = (const float*)((const char*)ws + 5 * 32 * 16 * sizeof(_Float16));
  v8f CI, CF, CG, CO, CZ;
  {
    float a = bia[m], b = bia[16 + m], c = bia[32 + m], d = bia[48 + m], e = bia[64 + m];
#pragma unroll
    for (int r = 0; r < 8; ++r) { CI[r] = a; CF[r] = b; CG[r] = c; CO[r] = d; CZ[r] = e; }
  }

  // zero h scratch
#pragma unroll
  for (int k = 0; k < HS_WAVE / 32; ++k) hs[w][lane + 32 * k] = 0.0f;

  const int xoff   = m * XS_ROW + hi * (8 * T_CHUNK);  // + j*T_CHUNK + t
  const int hload  = m * HS_ROW + hi * 8;
  const int hstore = hi * 8;

  v8f c_state;
#pragma unroll
  for (int r = 0; r < 8; ++r) c_state[r] = 0.0f;

#ifdef HAVE_TDM
  const unsigned lds0 = (unsigned)(uintptr_t)&xs[w][0][0];
  const unsigned lds1 = (unsigned)(uintptr_t)&xs[w][1][0];
  const unsigned long long gbase =
      (unsigned long long)(uintptr_t)x + (unsigned long long)b0 * (C_CH * T_SEQ * 4ull);
  tdm_issue(lds0, gbase);                    // chunk 0 -> buf 0
#endif

  int buf = 0;
#pragma unroll 1
  for (int cc = 0; cc < N_CHUNK; ++cc) {
#ifdef HAVE_TDM
    if (cc + 1 < N_CHUNK) {                  // prefetch next chunk -> other buf
      tdm_issue(buf ? lds0 : lds1, gbase + (unsigned long long)(cc + 1) * (T_CHUNK * 4ull));
      tdm_wait1();                           // oldest (current buf) complete
    } else {
      tdm_wait0();
    }
#else
    // Fallback: manual staged copy into xs[w][buf] as [m][c][t]
    const int t0 = cc * T_CHUNK;
#pragma unroll 1
    for (int it = 0; it < (ROWS_W * C_CH * T_CHUNK) / 32; ++it) {
      int e  = it * 32 + lane;
      int t  = e % T_CHUNK;
      int p  = e / T_CHUNK;
      int c  = p % C_CH;
      int mm = p / C_CH;
      xs[w][buf][mm * XS_ROW + c * T_CHUNK + t] =
          x[(size_t)(b0 + mm) * (C_CH * T_SEQ) + (size_t)c * T_SEQ + t0 + t];
    }
#endif

    const float* xb = &xs[w][buf][xoff];
#pragma unroll 1
    for (int t = 0; t < T_CHUNK; ++t) {
      const float* hp = &hs[w][hload];
      v16h A;
#pragma unroll
      for (int j = 0; j < 8; ++j) A[j] = (_Float16)xb[j * T_CHUNK + t];  // x (+pad)
#pragma unroll
      for (int k = 0; k < 8; ++k) A[8 + k] = (_Float16)hp[k];            // h (+pad)

      v8f gi = __builtin_amdgcn_wmma_f32_16x16x32_f16(false, A, false, Bi, (short)0, CI, false, false);
      v8f gf = __builtin_amdgcn_wmma_f32_16x16x32_f16(false, A, false, Bf, (short)0, CF, false, false);
      v8f gg = __builtin_amdgcn_wmma_f32_16x16x32_f16(false, A, false, Bg, (short)0, CG, false, false);
      v8f go = __builtin_amdgcn_wmma_f32_16x16x32_f16(false, A, false, Bo, (short)0, CO, false, false);

      v8f h_reg;
#pragma unroll
      for (int r = 0; r < 8; ++r) {
        float iv = sigm(gi[r]);
        float fv = sigm(gf[r]);
        float gv = tanhx(gg[r]);
        float ov = sigm(go[r]);
        float cv = fv * c_state[r] + iv * gv;
        c_state[r] = cv;
        h_reg[r] = ov * tanhx(cv);
      }
      // Unpredicated: cols 12..15 receive bounded garbage that only ever
      // multiplies zero B rows (K=28..31 in-loop, K=12..15 in final proj).
#pragma unroll
      for (int r = 0; r < 8; ++r)
        hs[w][(hstore + r) * HS_ROW + m] = h_reg[r];
    }
    buf ^= 1;
  }

  // Final projection: out = hT @ W_out^T + b_out (one more WMMA)
  {
    const float* hp = &hs[w][hload];
    v16h A;
#pragma unroll
    for (int k = 0; k < 8; ++k) A[k] = (_Float16)hp[k];      // K 0..15 = h(+pad)
#pragma unroll
    for (int k = 0; k < 8; ++k) A[8 + k] = (_Float16)0.0f;   // K 16..31 = 0
    v8f oz = __builtin_amdgcn_wmma_f32_16x16x32_f16(false, A, false, Bz, (short)0, CZ, false, false);
    if (m < HID) {
#pragma unroll
      for (int r = 0; r < 8; ++r)
        out[(size_t)(b0 + hstore + r) * HID + m] = oz[r];
    }
  }
}

extern "C" void kernel_launch(void* const* d_in, const int* in_sizes, int n_in,
                              void* d_out, int out_size, void* d_ws, size_t ws_size,
                              hipStream_t stream) {
  (void)in_sizes; (void)n_in; (void)out_size; (void)ws_size;
  const float* x     = (const float*)d_in[0];
  const float* W_pre = (const float*)d_in[1];
  const float* b_pre = (const float*)d_in[2];
  const float* W_ih  = (const float*)d_in[3];
  const float* W_hh  = (const float*)d_in[4];
  const float* b_ih  = (const float*)d_in[5];
  const float* b_hh  = (const float*)d_in[6];
  const float* W_out = (const float*)d_in[7];
  const float* b_out = (const float*)d_in[8];

  lstm_prep<<<1, 192, 0, stream>>>(W_pre, b_pre, W_ih, W_hh, b_ih, b_hh, W_out, b_out, d_ws);
  lstm_main<<<B_TOT / (WAVES * ROWS_W), BLOCK, 0, stream>>>(x, d_ws, (float*)d_out);
}
